// Topological_loss_15796889715000
// MI455X (gfx1250) — compile-verified, matrix operand order checked by reading, not run on hardware
//
#include <hip/hip_runtime.h>
#include <stdint.h>

#define BATCH 8
#define IMG_H 128
#define IMG_W 128
#define NPIX (IMG_H * IMG_W)   // 16384
#define NIMG (2 * BATCH)       // 16 (8 pred + 8 gt)
#define KTOP 128
#define NPTS (2 * KTOP)        // 256
#define SINK_EPS 0.1f
#define SINK_ITERS 150
#define TOPO_LAMBDA 0.1f

typedef unsigned long long u64;
typedef unsigned int u32;

typedef __attribute__((ext_vector_type(2)))  float    v2f;
typedef __attribute__((ext_vector_type(8)))  float    v8f;
typedef __attribute__((ext_vector_type(16))) _Float16 v16h;

// ---------------------------------------------------------------- utilities
__device__ __forceinline__ u32 flip_f32(float f) {
  // monotone float -> uint map (ascending)
  u32 u = __float_as_uint(f);
  return (u & 0x80000000u) ? ~u : (u | 0x80000000u);
}

// ------------------------------------------------------------------- init
__global__ void k_init(const float* __restrict__ pred, const float* __restrict__ gt,
                       u64* __restrict__ sortedK, u64* __restrict__ persKey,
                       int* __restrict__ parent, int* __restrict__ pairB,
                       int* __restrict__ pairD) {
  int idx = blockIdx.x * blockDim.x + threadIdx.x;
  if (idx >= NIMG * NPIX) return;
  int g = idx / NPIX, pix = idx % NPIX;
  const float* src = (g < BATCH) ? pred + (size_t)g * NPIX : gt + (size_t)(g - BATCH) * NPIX;
  u32 fk = flip_f32(src[pix]);
  sortedK[idx] = ((u64)fk << 32) | (u32)pix;          // stable ascending argsort key
  persKey[idx] = 0xFFFFFFFF00000000ull | (u32)pix;    // padding pairs: sort last
  parent[idx] = -1;
  pairB[idx] = 0;
  pairD[idx] = 0;
}

// ---------------------------------------------------- in-block bitonic sort
// One block sorts one image's 16384 u64 keys ascending (in global memory).
__global__ void k_bitonic(u64* __restrict__ data) {
  u64* a = data + (size_t)blockIdx.x * NPIX;
  const int T = blockDim.x;
  for (int kk = 2; kk <= NPIX; kk <<= 1) {
    for (int j = kk >> 1; j > 0; j >>= 1) {
      for (int i = threadIdx.x; i < NPIX; i += T) {
        int l = i ^ j;
        if (l > i) {
          u64 x = a[i], y = a[l];
          bool up = ((i & kk) == 0);
          if (up ? (x > y) : (x < y)) { a[i] = y; a[l] = x; }
        }
      }
      __syncthreads();
    }
  }
}

// ------------------------------------------------------------------- rank
__global__ void k_rank(const u64* __restrict__ sortedK, int* __restrict__ rankA) {
  int idx = blockIdx.x * blockDim.x + threadIdx.x;
  if (idx >= NIMG * NPIX) return;
  int g = idx / NPIX, pos = idx % NPIX;
  int pix = (int)(sortedK[idx] & 0xFFFFFFFFull);
  rankA[g * NPIX + pix] = pos;
}

// -------------------------------------------------------------- union-find
__device__ __forceinline__ int uf_find(int* par, int i) {
  int r = i;
  while (par[r] != r) r = par[r];
  while (par[i] != r) { int nx = par[i]; par[i] = r; i = nx; }
  return r;
}

// One block per image; lane 0 runs the elder-rule sweep serially.
__global__ void k_unionfind(const float* __restrict__ pred, const float* __restrict__ gt,
                            const u64* __restrict__ sortedK, const int* __restrict__ rankA,
                            int* __restrict__ parent, int* __restrict__ rb,
                            int* __restrict__ pairB, int* __restrict__ pairD,
                            u64* __restrict__ persKey) {
  if (threadIdx.x != 0) return;
  int g = blockIdx.x;
  const float* f = (g < BATCH) ? pred + (size_t)g * NPIX : gt + (size_t)(g - BATCH) * NPIX;
  const u64* ord = sortedK + (size_t)g * NPIX;
  const int* rk = rankA + (size_t)g * NPIX;
  int* par = parent + (size_t)g * NPIX;
  int* rbg = rb + (size_t)g * NPIX;
  int* pB = pairB + (size_t)g * NPIX;
  int* pD = pairD + (size_t)g * NPIX;
  u64* pk = persKey + (size_t)g * NPIX;

  int cnt = 0;
  for (int s = 0; s < NPIX; ++s) {
    int p = (int)(ord[s] & 0xFFFFFFFFull);
    par[p] = p;
    rbg[p] = p;
    int y = p / IMG_W, x = p % IMG_W;
    int nb0 = (y > 0) ? p - IMG_W : -1;
    int nb1 = (y < IMG_H - 1) ? p + IMG_W : -1;
    int nb2 = (x > 0) ? p - 1 : -1;
    int nb3 = (x < IMG_W - 1) ? p + 1 : -1;
    int nbs[4] = {nb0, nb1, nb2, nb3};
    for (int t = 0; t < 4; ++t) {
      int qi = nbs[t];
      if (qi >= 0 && par[qi] != -1) {
        int rp = uf_find(par, p);
        int rq = uf_find(par, qi);
        if (rp != rq) {
          int bpix;
          if (rk[rbg[rp]] <= rk[rbg[rq]]) { bpix = rbg[rq]; par[rq] = rp; }
          else                           { bpix = rbg[rp]; par[rp] = rq; }
          float pers = f[p] - f[bpix];
          pB[cnt] = bpix;
          pD[cnt] = p;
          pk[cnt] = ((u64)(~flip_f32(pers)) << 32) | (u32)cnt;  // descending pers, stable
          ++cnt;
        }
      }
    }
  }
  // essential feature: global min paired with global max
  int b0 = (int)(ord[0] & 0xFFFFFFFFull);
  int dN = (int)(ord[NPIX - 1] & 0xFFFFFFFFull);
  float pers = f[dN] - f[b0];
  pB[cnt] = b0;
  pD[cnt] = dN;
  pk[cnt] = ((u64)(~flip_f32(pers)) << 32) | (u32)cnt;
}

// -------------------------------------------------------------------- topK
__global__ void k_topk(const u64* __restrict__ persKey, const int* __restrict__ pairB,
                       const int* __restrict__ pairD, int* __restrict__ topB,
                       int* __restrict__ topD) {
  int idx = blockIdx.x * blockDim.x + threadIdx.x;
  if (idx >= NIMG * KTOP) return;
  int g = idx / KTOP, k = idx % KTOP;
  int slot = (int)(persKey[(size_t)g * NPIX + k] & 0xFFFFFFFFull);
  topB[idx] = pairB[(size_t)g * NPIX + slot];
  topD[idx] = pairD[(size_t)g * NPIX + slot];
}

// ------------------------------------------------- cost matrix via f32 WMMA
// C = ||x||^2 + ||y||^2 - 2 x.y,  Gram term done on the matrix core.
__global__ void k_buildC(const float* __restrict__ pred, const float* __restrict__ gt,
                         const int* __restrict__ topB, const int* __restrict__ topD,
                         float* __restrict__ Cmat, float* __restrict__ Ctmat) {
  int b = blockIdx.x;
  int t = threadIdx.x;  // 0..255 -> 8 wave32
  __shared__ float Xb[NPTS], Xd[NPTS], Yb[NPTS], Yd[NPTS], nX[NPTS], nY[NPTS];

  const float* fp = pred + (size_t)b * NPIX;
  const float* fg = gt + (size_t)b * NPIX;
  if (t < KTOP) {
    int gp = b, gg = BATCH + b;
    float pb = fp[topB[gp * KTOP + t]];
    float pd = fp[topD[gp * KTOP + t]];
    float qb = fg[topB[gg * KTOP + t]];
    float qd = fg[topD[gg * KTOP + t]];
    Xb[t] = pb;        Xd[t] = pd;        // Dp
    Yb[t] = qb;        Yd[t] = qd;        // Dg
    float mg = 0.5f * (qb + qd);
    Xb[KTOP + t] = mg; Xd[KTOP + t] = mg; // diag_proj(Dg)
    float mp = 0.5f * (pb + pd);
    Yb[KTOP + t] = mp; Yd[KTOP + t] = mp; // diag_proj(Dp)
  }
  __syncthreads();
  nX[t] = Xb[t] * Xb[t] + Xd[t] * Xd[t];
  nY[t] = Yb[t] * Yb[t] + Yd[t] * Yd[t];
  __syncthreads();

  int wave = t >> 5, lane = t & 31;
  int l2 = lane & 15, hi = lane >> 4;
  float* C = Cmat + (size_t)b * NPTS * NPTS;
  float* Ct = Ctmat + (size_t)b * NPTS * NPTS;

  for (int tile = wave; tile < 256; tile += 8) {  // uniform per wave: EXEC all ones
    int ti = tile >> 4, tj = tile & 15;
#if __has_builtin(__builtin_amdgcn_wmma_f32_16x16x4_f32)
    // A 16x4 f32: lanes 0-15 hold (M=l2, K=0,1); lanes 16-31 hold K=2,3 (zero pad)
    v2f av = (v2f)0.0f, bv = (v2f)0.0f;
    if (!hi) {
      av.x = Xb[ti * 16 + l2];
      av.y = Xd[ti * 16 + l2];
      bv.x = Yb[tj * 16 + l2];
      bv.y = Yd[tj * 16 + l2];
    }
    v8f c = (v8f)0.0f;
    v8f dm = __builtin_amdgcn_wmma_f32_16x16x4_f32(false, av, false, bv,
                                                   (short)0, c, false, false);
#else
    // fallback: codegen-confirmed f16 WMMA, K padded 2->32 with zeros
    v16h av = (v16h)(_Float16)0.0f, bv = (v16h)(_Float16)0.0f;
    if (!hi) {
      av[0] = (_Float16)Xb[ti * 16 + l2];
      av[1] = (_Float16)Xd[ti * 16 + l2];
      bv[0] = (_Float16)Yb[tj * 16 + l2];
      bv[1] = (_Float16)Yd[tj * 16 + l2];
    }
    v8f c = (v8f)0.0f;
    v8f dm = __builtin_amdgcn_wmma_f32_16x16x32_f16(false, av, false, bv,
                                                    (short)0, c, false, false);
#endif
    for (int r = 0; r < 8; ++r) {
      int m = ti * 16 + hi * 8 + r;  // C/D layout: VGPR r -> M = r (+8 for hi lanes)
      int nn = tj * 16 + l2;
      float cost = nX[m] + nY[nn] - 2.0f * dm[r];
      if (m >= KTOP && nn >= KTOP) cost = 0.0f;  // diagonal<->diagonal is free
      cost = fmaxf(cost, 0.0f);
      C[(size_t)m * NPTS + nn] = cost;
      Ct[(size_t)nn * NPTS + m] = cost;
    }
  }
}

// --------------------------------------------------------- Sinkhorn (log dom)
__global__ void k_sinkhorn(const float* __restrict__ Cmat, const float* __restrict__ Ctmat,
                           float* __restrict__ wout) {
  int b = blockIdx.x, t = threadIdx.x;  // 256 threads = NPTS rows
  const float* C = Cmat + (size_t)b * NPTS * NPTS;
  const float* Ct = Ctmat + (size_t)b * NPTS * NPTS;
  __shared__ float fs[NPTS], gs[NPTS], red[NPTS];
  const float invE = 1.0f / SINK_EPS;
  const float loga = -logf((float)NPTS);

  fs[t] = 0.0f;
  gs[t] = 0.0f;
  __syncthreads();

  const float* Crow = C + (size_t)t * NPTS;
  const float* Ctrow = Ct + (size_t)t * NPTS;
  __builtin_prefetch(Crow, 0, 3);
  __builtin_prefetch(Ctrow, 0, 3);

  for (int it = 0; it < SINK_ITERS; ++it) {
    // f_i = -eps * ( LSE_j((g_j - C_ij)/eps) + loga )
    float m = -3.0e38f;
    for (int j = 0; j < NPTS; ++j) m = fmaxf(m, (gs[j] - Crow[j]) * invE);
    float s = 0.0f;
    for (int j = 0; j < NPTS; ++j) s += expf((gs[j] - Crow[j]) * invE - m);
    float fi = -SINK_EPS * (m + logf(s) + loga);
    __syncthreads();
    fs[t] = fi;
    __syncthreads();
    // g_j = -eps * ( LSE_i((f_i - C_ij)/eps) + loga )
    m = -3.0e38f;
    for (int i = 0; i < NPTS; ++i) m = fmaxf(m, (fs[i] - Ctrow[i]) * invE);
    s = 0.0f;
    for (int i = 0; i < NPTS; ++i) s += expf((fs[i] - Ctrow[i]) * invE - m);
    float gj = -SINK_EPS * (m + logf(s) + loga);
    __syncthreads();
    gs[t] = gj;
    __syncthreads();
  }

  // total = sum P*C * n ; w = sqrt(total)
  float acc = 0.0f;
  float fi = fs[t];
  for (int j = 0; j < NPTS; ++j) {
    float cc = Crow[j];
    acc += expf((fi + gs[j] - cc) * invE + 2.0f * loga) * cc;
  }
  red[t] = acc;
  __syncthreads();
  for (int off = NPTS / 2; off > 0; off >>= 1) {
    if (t < off) red[t] += red[t + off];
    __syncthreads();
  }
  if (t == 0) wout[b] = sqrtf(fmaxf(red[0] * (float)NPTS, 0.0f));
}

// ------------------------------------------------------------------- final
__global__ void k_final(const float* __restrict__ wout, float* __restrict__ out) {
  if (blockIdx.x == 0 && threadIdx.x == 0) {
    float s = 0.0f;
    for (int i = 0; i < BATCH; ++i) s += wout[i];
    out[0] = TOPO_LAMBDA * (s / (float)BATCH);
  }
}

// ------------------------------------------------------------------ launch
extern "C" void kernel_launch(void* const* d_in, const int* in_sizes, int n_in,
                              void* d_out, int out_size, void* d_ws, size_t ws_size,
                              hipStream_t stream) {
  (void)in_sizes; (void)n_in; (void)out_size; (void)ws_size;
  const float* pred = (const float*)d_in[0];
  const float* gt = (const float*)d_in[1];
  float* out = (float*)d_out;

  char* ws = (char*)d_ws;
  size_t o = 0;
  u64* sortedK = (u64*)(ws + o); o += (size_t)NIMG * NPIX * 8;
  u64* persKey = (u64*)(ws + o); o += (size_t)NIMG * NPIX * 8;
  int* rankA   = (int*)(ws + o); o += (size_t)NIMG * NPIX * 4;
  int* parent  = (int*)(ws + o); o += (size_t)NIMG * NPIX * 4;
  int* rb      = (int*)(ws + o); o += (size_t)NIMG * NPIX * 4;
  int* pairB   = (int*)(ws + o); o += (size_t)NIMG * NPIX * 4;
  int* pairD   = (int*)(ws + o); o += (size_t)NIMG * NPIX * 4;
  int* topB    = (int*)(ws + o); o += (size_t)NIMG * KTOP * 4;
  int* topD    = (int*)(ws + o); o += (size_t)NIMG * KTOP * 4;
  float* Cmat  = (float*)(ws + o); o += (size_t)BATCH * NPTS * NPTS * 4;
  float* Ctmat = (float*)(ws + o); o += (size_t)BATCH * NPTS * NPTS * 4;
  float* wout  = (float*)(ws + o); o += (size_t)BATCH * 4;

  k_init<<<(NIMG * NPIX + 255) / 256, 256, 0, stream>>>(pred, gt, sortedK, persKey,
                                                        parent, pairB, pairD);
  k_bitonic<<<NIMG, 1024, 0, stream>>>(sortedK);
  k_rank<<<(NIMG * NPIX + 255) / 256, 256, 0, stream>>>(sortedK, rankA);
  k_unionfind<<<NIMG, 32, 0, stream>>>(pred, gt, sortedK, rankA, parent, rb,
                                       pairB, pairD, persKey);
  k_bitonic<<<NIMG, 1024, 0, stream>>>(persKey);
  k_topk<<<(NIMG * KTOP + 255) / 256, 256, 0, stream>>>(persKey, pairB, pairD, topB, topD);
  k_buildC<<<BATCH, 256, 0, stream>>>(pred, gt, topB, topD, Cmat, Ctmat);
  k_sinkhorn<<<BATCH, 256, 0, stream>>>(Cmat, Ctmat, wout);
  k_final<<<1, 64, 0, stream>>>(wout, out);
}
